// Correlation_29609504538974
// MI455X (gfx1250) — matrix operand order repaired and checked by reading, not compiled, not run on hardware
//
#include <hip/hip_runtime.h>

// ---------------------------------------------------------------------------
// Multi-scale spatial correlation (patch 3x3) for MI455X / gfx1250.
//
// out[b, di*3+dj, y, x] = leaky_relu( (1/C) * sum_c f1[b,c,y,x] *
//                                     f2[b,c, y+di-1, x+dj-1], 0.01 )
//
// Banded-Gram WMMA formulation (memory-bound: ~420 MB traffic, ~18us floor
// at 23.3 TB/s; bf16 split-WMMA work ~21 GFLOP, far below the bf16 matrix
// roofline):
//   A = 16 feat1 pixels of one output row (M), K = 32 channels (bf16)
//   B = 16 feat2 pixels of row y+di-1, shifted -1 in x (N)
//   D[m,n] = <f1(x0+m), f2(x0+n-1)> ; diagonals n-m in {0,1,2} = dj -1,0,+1
// Tiles stride by 14 output pixels. f32 -> bf16 via truncating hi/lo split:
// hi*hi + hi*lo + lo*hi (lo*lo dropped, ~3e-5 relative error).
//
// Boundary handling without per-load masks: all addresses are clamped
// in-bounds, invalid A rows produce D rows that are never stored, and an
// invalid B column corrupts only this lane's own D column, whose outputs are
// by zero-padding semantics exactly leaky_relu(0)=0 -> substituted at store.
// ---------------------------------------------------------------------------

typedef __attribute__((ext_vector_type(16))) __bf16    v16bf;
typedef __attribute__((ext_vector_type(8)))  float     v8f;
typedef __attribute__((ext_vector_type(8)))  unsigned  v8u;

struct Packed2 { unsigned hi; unsigned lo; };

// Pack two f32 into one VGPR holding {bf16(a) | bf16(b)<<16} (truncating),
// and the same for the exact residuals a-hi(a), b-hi(b). v_perm_b32 packs
// the two high halfwords in a single instruction.
__device__ __forceinline__ Packed2 split_pack(float a, float b) {
    const unsigned ua = __builtin_bit_cast(unsigned, a);
    const unsigned ub = __builtin_bit_cast(unsigned, b);
    const unsigned ha = ua & 0xFFFF0000u;
    const unsigned hb = ub & 0xFFFF0000u;
    const float la = a - __builtin_bit_cast(float, ha);   // exact in f32
    const float lb = b - __builtin_bit_cast(float, hb);
    Packed2 r;
    r.hi = __builtin_amdgcn_perm(ub, ua, 0x07060302u);
    r.lo = __builtin_amdgcn_perm(__builtin_bit_cast(unsigned, lb),
                                 __builtin_bit_cast(unsigned, la), 0x07060302u);
    return r;
}

template<int C, int H, int W>
__global__ __launch_bounds__(32)
void corr_wmma_kernel(const float* __restrict__ f1,   // feat1 = x (current)
                      const float* __restrict__ f2,   // feat2 = x_pre
                      float* __restrict__ out)
{
    constexpr int   HW     = H * W;
    constexpr int   tilesX = (W + 13) / 14;
    constexpr int   nChunk = C / 32;
    constexpr float scale  = 1.0f / (float)C;

    const int lane = threadIdx.x;          // one wave32 per block
    const int nOrM = lane & 15;            // A row index m / B col index n
    const int half = lane >> 4;

    int t = blockIdx.x;
    const int tx = t % tilesX;  t /= tilesX;
    const int y  = t % H;
    const int b  = t / H;

    const int x0 = tx * 14;                // 14 owned output pixels per tile

    // A (16x32 bf16) VGPR layout per ISA: lane half h, slot s:
    //   k = 8*h + {0..7 for s=0..7, 16..23 for s=8..15}
    constexpr int fA[16] = {0,1,2,3,4,5,6,7,16,17,18,19,20,21,22,23};

    // ---- feat1 (A): pixel x0+m of row y (clamped; bad rows never stored) ----
    const int xa  = x0 + nOrM;
    const int xaC = (xa < W) ? xa : (W - 1);
    const float* pA = f1 + (b * C + 8 * half) * HW + y * W + xaC;

    // ---- feat2 (B): pixel x0-1+n of row y+di-1 (clamped) ----
    // B (32x16 bf16): lane half h, slot s -> same K order as A (fA table)
    const int  xb   = x0 - 1 + nOrM;
    const int  xbC  = (xb < 0) ? 0 : ((xb >= W) ? (W - 1) : xb);
    const bool xbOk = (xb >= 0 && xb < W);

    bool valid[3];
    const float* pB[3];
#pragma unroll
    for (int di = 0; di < 3; ++di) {
        const int yb = y + di - 1;
        valid[di] = (yb >= 0 && yb < H) && xbOk;   // zero-pad -> store 0 later
        const int ybC = (yb < 0) ? 0 : ((yb >= H) ? (H - 1) : yb);
        pB[di] = f2 + (b * C + 8 * half) * HW + ybC * W + xbC;
    }

    v8f acc[3] = {};                       // f32 16x16 accumulators, one per di

    for (int kc = 0; kc < nChunk; ++kc) {
        if (kc + 1 < nChunk)               // pull next A chunk toward the WGP
            __builtin_prefetch(pA + 32 * HW, 0, 0);

        // ---- issue all 64 loads up front (independent immediate offsets) ----
        float af[16], bf[3][16];
#pragma unroll
        for (int s = 0; s < 16; ++s) af[s] = pA[fA[s] * HW];
#pragma unroll
        for (int di = 0; di < 3; ++di)
#pragma unroll
            for (int s = 0; s < 16; ++s) bf[di][s] = pB[di][fA[s] * HW];

        pA += 32 * HW;
#pragma unroll
        for (int di = 0; di < 3; ++di) pB[di] += 32 * HW;

        // ---- split + pack A ----
        v8u ahiU, aloU;
#pragma unroll
        for (int i = 0; i < 8; ++i) {
            const Packed2 p = split_pack(af[2 * i], af[2 * i + 1]);
            ahiU[i] = p.hi;
            aloU[i] = p.lo;
        }
        const v16bf ahi = __builtin_bit_cast(v16bf, ahiU);
        const v16bf alo = __builtin_bit_cast(v16bf, aloU);

        // ---- per-di: split + pack B, 3 split-product WMMAs ----
#pragma unroll
        for (int di = 0; di < 3; ++di) {
            v8u bhiU, bloU;
#pragma unroll
            for (int i = 0; i < 8; ++i) {
                const Packed2 p = split_pack(bf[di][2 * i], bf[di][2 * i + 1]);
                bhiU[i] = p.hi;
                bloU[i] = p.lo;
            }
            const v16bf bhi = __builtin_bit_cast(v16bf, bhiU);
            const v16bf blo = __builtin_bit_cast(v16bf, bloU);

            acc[di] = __builtin_amdgcn_wmma_f32_16x16x32_bf16(
                          false, ahi, false, bhi, (short)0, acc[di], false, false);
            acc[di] = __builtin_amdgcn_wmma_f32_16x16x32_bf16(
                          false, ahi, false, blo, (short)0, acc[di], false, false);
            acc[di] = __builtin_amdgcn_wmma_f32_16x16x32_bf16(
                          false, alo, false, bhi, (short)0, acc[di], false, false);
        }
    }

    // ---- harvest diagonals: D layout lane = n + 16*(m/8), vgpr r = m%8 ----
#pragma unroll
    for (int di = 0; di < 3; ++di) {
#pragma unroll
        for (int r = 0; r < 8; ++r) {
            const int m = r + 8 * half;        // output pixel within tile
            const int d = nOrM - m;            // dj + 1, valid in [0,2]
            const int x = x0 + m;
            if (d >= 0 && d <= 2 && m < 14 && x < W) {
                float v = valid[di] ? (acc[di][r] * scale) : 0.0f;
                v = (v > 0.0f) ? v : 0.01f * v;          // leaky ReLU
                out[((b * 9 + di * 3 + d) * H + y) * W + x] = v;
            }
        }
    }
}

template<int C, int H, int W>
static inline void launch_level(const float* f1, const float* f2, float* out,
                                hipStream_t stream) {
    constexpr int B = 16;
    constexpr int tilesX = (W + 13) / 14;
    dim3 grid((unsigned)(B * H * tilesX));
    corr_wmma_kernel<C, H, W><<<grid, 32, 0, stream>>>(f1, f2, out);
}

extern "C" void kernel_launch(void* const* d_in, const int* in_sizes, int n_in,
                              void* d_out, int out_size, void* d_ws, size_t ws_size,
                              hipStream_t stream) {
    // d_in[0..5] = x_pre{0..5} (feat2), d_in[6..11] = x{0..5} (feat1)
    float* out = (float*)d_out;
    const int B = 16;

    launch_level< 512, 64, 64>((const float*)d_in[6],  (const float*)d_in[0], out, stream);
    out += (size_t)B * 9 * 64 * 64;
    launch_level<1024, 32, 32>((const float*)d_in[7],  (const float*)d_in[1], out, stream);
    out += (size_t)B * 9 * 32 * 32;
    launch_level< 512, 16, 16>((const float*)d_in[8],  (const float*)d_in[2], out, stream);
    out += (size_t)B * 9 * 16 * 16;
    launch_level< 256,  8,  8>((const float*)d_in[9],  (const float*)d_in[3], out, stream);
    out += (size_t)B * 9 * 8 * 8;
    launch_level< 256,  4,  4>((const float*)d_in[10], (const float*)d_in[4], out, stream);
    out += (size_t)B * 9 * 4 * 4;
    launch_level< 256,  2,  2>((const float*)d_in[11], (const float*)d_in[5], out, stream);
}
